// DecoderLayer_88424786690751
// MI455X (gfx1250) — matrix operand order repaired and checked by reading, not compiled, not run on hardware
//
#include <hip/hip_runtime.h>
#include <hip/hip_bf16.h>
#include <stdint.h>

typedef unsigned short u16;
typedef __attribute__((ext_vector_type(16))) __bf16 v16bf;
typedef __attribute__((ext_vector_type(8)))  float  v8f;

constexpr int kH   = 768;
constexpr int kNH  = 12;
constexpr int kHD  = 64;
constexpr int kI   = 2048;
constexpr int kE   = 8;
constexpr int kSeq = 2048;
constexpr int kB   = 2;
constexpr int kNT  = kB * kSeq;   // 4096 tokens

union Frag { uint4 u[2]; v16bf bf; };

__device__ inline u16 f2bf(float f) {
  uint32_t u = __float_as_uint(f);
  u += 0x7FFFu + ((u >> 16) & 1u);      // round-to-nearest-even
  return (u16)(u >> 16);
}
__device__ inline float bf2f(u16 h) { return __uint_as_float(((uint32_t)h) << 16); }

__device__ inline float fast_rcp(float x) { return __builtin_amdgcn_rcpf(x); }

__device__ inline v8f wmma_bf16(v16bf a, v16bf b, v8f c) {
  return __builtin_amdgcn_wmma_f32_16x16x32_bf16(false, a, false, b, (short)0, c, false, false);
}

// CDNA5 async global->LDS copy (ASYNCcnt-tracked, bypasses VGPRs).
// LDS address operand = low 32 bits of the flat pointer (ISA aperture rule).
__device__ inline void async_ld_b128(void* lds, const void* gmem) {
  asm volatile("global_load_async_to_lds_b128 %0, %1, off"
               :: "v"((uint32_t)(uintptr_t)lds), "v"((uint64_t)(uintptr_t)gmem)
               : "memory");
}
// Wait until this wave's outstanding async ops <= N. Async loads complete in
// order, so N = ops issued for the *next* tile leaves exactly it in flight.
template <int N>
__device__ inline void wait_async() {
  asm volatile("s_wait_asynccnt %0" :: "i"(N) : "memory");
}

// ---------------------------------------------------------------- One-time
// weight prep: fp32 [K,N] -> bf16 transposed [N,K] via LDS tile (coalesced
// both ways). Runs once per launch; cost ~bytes/23.3TB/s, trivial vs GEMMs.
__global__ __launch_bounds__(256) void transpose_bf16_k(const float* __restrict__ W,
                                                        u16* __restrict__ WT,
                                                        int K, int N) {
  __shared__ u16 tile[64][65];
  const int k0 = blockIdx.x * 64, n0 = blockIdx.y * 64;
  for (int i = threadIdx.x; i < 64 * 64; i += 256) {
    const int r = i >> 6, c = i & 63;             // r = k offset, c = n offset
    tile[c][r] = f2bf(W[(size_t)(k0 + r) * N + n0 + c]);
  }
  __syncthreads();
  for (int i = threadIdx.x; i < 64 * 64; i += 256) {
    const int r = i >> 6, c = i & 63;             // r = n offset, c = k offset
    WT[(size_t)(n0 + r) * K + k0 + c] = tile[r][c];
  }
}

// ---------------------------------------------------------------- RMSNorm ->
// bf16 normalized activations. One block per token row.
__global__ __launch_bounds__(256) void rmsnorm_k(const float* __restrict__ x,
                                                 const float* __restrict__ w,
                                                 u16* __restrict__ out) {
  const int row = blockIdx.x;
  const float* xr = x + (size_t)row * kH;
  float s = 0.f;
  for (int i = threadIdx.x; i < kH; i += 256) { float v = xr[i]; s += v * v; }
  __shared__ float red[256];
  red[threadIdx.x] = s; __syncthreads();
  for (int st = 128; st > 0; st >>= 1) {
    if (threadIdx.x < st) red[threadIdx.x] += red[threadIdx.x + st];
    __syncthreads();
  }
  const float rs = rsqrtf(red[0] * (1.f / kH) + 1e-6f);
  for (int i = threadIdx.x; i < kH; i += 256)
    out[(size_t)row * kH + i] = f2bf(w[i] * xr[i] * rs);
}

// ---------------------------------------------------------------- Generic
// bf16 WMMA GEMM: C[M,N] = A[M,K](bf16) x BT[N,K](bf16, pre-transposed).
// Block tile 128x64, BK=32, 8 waves x 32x32 tiles. Double-buffered LDS with
// async-to-LDS DMA for both A and B; inner loop = async issue + ds_load + wmma.
// MODE 0: store bf16            MODE 1: f32 out = acc + resid[row,col]
// MODE 3: f32 out += aux[row*8] * acc   (MoE down accumulate; aux = comb+e)
template <int MODE>
__global__ __launch_bounds__(256) void gemm_k(const u16* __restrict__ A,
                                              const u16* __restrict__ BT,
                                              u16* __restrict__ outb,
                                              float* __restrict__ outf,
                                              const float* __restrict__ aux,
                                              int M, int N, int K) {
  __shared__ __align__(16) u16 sA[2][128 * 32];
  __shared__ __align__(16) u16 sBT[2][64 * 32];
  const int bm = blockIdx.x * 128, bn = blockIdx.y * 64;
  const int wave = threadIdx.x >> 5, lane = threadIdx.x & 31;
  const int hf = lane >> 4, ln = lane & 15;
  const int wr = wave & 3, wc = wave >> 2;
  const v8f zz = {};
  v8f acc[2][2];
#pragma unroll
  for (int i = 0; i < 2; i++)
#pragma unroll
    for (int j = 0; j < 2; j++) acc[i][j] = zz;

  auto stage = [&](int kt, int buf) {
    const int k0 = kt * 32;
#pragma unroll
    for (int i = 0; i < 2; i++) {                 // A: 2 x b128 per thread
      const int e = (threadIdx.x + i * 256) * 8;
      const int row = e >> 5, kk = e & 31;
      async_ld_b128(&sA[buf][e], &A[(size_t)(bm + row) * K + k0 + kk]);
    }
    {                                             // B: 1 x b128 per thread
      const int n = threadIdx.x >> 2, ko = (threadIdx.x & 3) * 8;
      async_ld_b128(&sBT[buf][n * 32 + ko], &BT[(size_t)(bn + n) * K + k0 + ko]);
    }
  };

  const int nk = K >> 5;
  stage(0, 0);
  for (int kt = 0; kt < nk; ++kt) {
    const int cur = kt & 1;
    if (kt + 1 < nk) { stage(kt + 1, cur ^ 1); wait_async<3>(); }
    else wait_async<0>();
    __syncthreads();

    Frag af[2], bg[2];
#pragma unroll
    for (int i = 0; i < 2; i++) {
      const u16* ap = &sA[cur][(wr * 32 + i * 16 + ln) * 32 + 8 * hf];
      af[i].u[0] = *(const uint4*)ap;
      af[i].u[1] = *(const uint4*)(ap + 16);
    }
#pragma unroll
    for (int j = 0; j < 2; j++) {
      const u16* bp = &sBT[cur][(wc * 32 + j * 16 + ln) * 32 + 8 * hf];
      bg[j].u[0] = *(const uint4*)bp;
      bg[j].u[1] = *(const uint4*)(bp + 16);
    }
#pragma unroll
    for (int i = 0; i < 2; i++)
#pragma unroll
      for (int j = 0; j < 2; j++)
        acc[i][j] = wmma_bf16(af[i].bf, bg[j].bf, acc[i][j]);
    __syncthreads();
  }

#pragma unroll
  for (int i = 0; i < 2; i++)
#pragma unroll
    for (int j = 0; j < 2; j++)
#pragma unroll
      for (int r = 0; r < 8; r++) {
        const int row = bm + wr * 32 + i * 16 + r + 8 * hf;
        const int col = bn + wc * 32 + j * 16 + ln;
        const float v = acc[i][j][r];
        const size_t idx = (size_t)row * N + col;
        if (MODE == 0) outb[idx] = f2bf(v);
        else if (MODE == 1) outf[idx] = v + aux[idx];
        else if (MODE == 3) outf[idx] += aux[(size_t)row * 8] * v;
      }
}

// ---------------------------------------------------------------- Fused
// gate/up GEMM with SiLU: h = silu(A@Bg) * (A@Bu); Bg/Bu pre-transposed bf16.
__global__ __launch_bounds__(256) void gemm_gated_k(const u16* __restrict__ A,
                                                    const u16* __restrict__ BgT,
                                                    const u16* __restrict__ BuT,
                                                    u16* __restrict__ outb,
                                                    int M, int N, int K) {
  __shared__ __align__(16) u16 sA[2][128 * 32];
  __shared__ __align__(16) u16 sG[2][64 * 32];
  __shared__ __align__(16) u16 sU[2][64 * 32];
  const int bm = blockIdx.x * 128, bn = blockIdx.y * 64;
  const int wave = threadIdx.x >> 5, lane = threadIdx.x & 31;
  const int hf = lane >> 4, ln = lane & 15;
  const int wr = wave & 3, wc = wave >> 2;
  const v8f zz = {};
  v8f ag[2][2], au[2][2];
#pragma unroll
  for (int i = 0; i < 2; i++)
#pragma unroll
    for (int j = 0; j < 2; j++) { ag[i][j] = zz; au[i][j] = zz; }

  auto stage = [&](int kt, int buf) {
    const int k0 = kt * 32;
#pragma unroll
    for (int i = 0; i < 2; i++) {
      const int e = (threadIdx.x + i * 256) * 8;
      const int row = e >> 5, kk = e & 31;
      async_ld_b128(&sA[buf][e], &A[(size_t)(bm + row) * K + k0 + kk]);
    }
    const int n = threadIdx.x >> 2, ko = (threadIdx.x & 3) * 8;
    async_ld_b128(&sG[buf][n * 32 + ko], &BgT[(size_t)(bn + n) * K + k0 + ko]);
    async_ld_b128(&sU[buf][n * 32 + ko], &BuT[(size_t)(bn + n) * K + k0 + ko]);
  };

  const int nk = K >> 5;
  stage(0, 0);
  for (int kt = 0; kt < nk; ++kt) {
    const int cur = kt & 1;
    if (kt + 1 < nk) { stage(kt + 1, cur ^ 1); wait_async<4>(); }
    else wait_async<0>();
    __syncthreads();

    Frag af[2], gf[2], uf[2];
#pragma unroll
    for (int i = 0; i < 2; i++) {
      const u16* ap = &sA[cur][(wr * 32 + i * 16 + ln) * 32 + 8 * hf];
      af[i].u[0] = *(const uint4*)ap;
      af[i].u[1] = *(const uint4*)(ap + 16);
    }
#pragma unroll
    for (int j = 0; j < 2; j++) {
      const int off = (wc * 32 + j * 16 + ln) * 32 + 8 * hf;
      gf[j].u[0] = *(const uint4*)&sG[cur][off];
      gf[j].u[1] = *(const uint4*)&sG[cur][off + 16];
      uf[j].u[0] = *(const uint4*)&sU[cur][off];
      uf[j].u[1] = *(const uint4*)&sU[cur][off + 16];
    }
#pragma unroll
    for (int i = 0; i < 2; i++)
#pragma unroll
      for (int j = 0; j < 2; j++) {
        ag[i][j] = wmma_bf16(af[i].bf, gf[j].bf, ag[i][j]);
        au[i][j] = wmma_bf16(af[i].bf, uf[j].bf, au[i][j]);
      }
    __syncthreads();
  }

#pragma unroll
  for (int i = 0; i < 2; i++)
#pragma unroll
    for (int j = 0; j < 2; j++)
#pragma unroll
      for (int r = 0; r < 8; r++) {
        const int row = bm + wr * 32 + i * 16 + r + 8 * hf;
        const int col = bn + wc * 32 + j * 16 + ln;
        const float g = ag[i][j][r];
        const float u = au[i][j][r];
        const float si = g * fast_rcp(1.f + __expf(-g));   // v_rcp_f32, no div chain
        outb[(size_t)row * N + col] = f2bf(si * u);
      }
}

// ---------------------------------------------------------------- RoPE +
// relayout token-major [tok][h*64+d] -> head-major [bh][s][d] (bf16).
__global__ __launch_bounds__(256) void rope_k(const u16* __restrict__ ql,
                                              const u16* __restrict__ kl,
                                              const u16* __restrict__ vl,
                                              u16* __restrict__ qa,
                                              u16* __restrict__ ka,
                                              u16* __restrict__ va) {
  const int gid = blockIdx.x * 256 + threadIdx.x;   // < kNT*kH
  const int tok = gid / kH, c = gid % kH;
  const int hh = c >> 6, d = c & 63;
  const int b = tok >> 11, s = tok & (kSeq - 1);
  const float inv = __powf(10000.f, -(float)(d & 31) * (1.f / 32.f));
  float cs, sn; __sincosf((float)s * inv, &cs, &sn);
  const size_t oidx = ((size_t)(b * kNH + hh) * kSeq + s) * kHD + d;
  const int other = (d < 32) ? (c + 32) : (c - 32);
  const float sg = (d < 32) ? -1.f : 1.f;
  const size_t base = (size_t)tok * kH;
  const float q1 = bf2f(ql[base + c]), q2 = bf2f(ql[base + other]);
  qa[oidx] = f2bf(q1 * cs + sg * q2 * sn);
  const float k1 = bf2f(kl[base + c]), k2 = bf2f(kl[base + other]);
  ka[oidx] = f2bf(k1 * cs + sg * k2 * sn);
  va[oidx] = vl[base + c];
}

// ---------------------------------------------------------------- Flash
// attention. Block = 128 q rows of one (b,h); 8 waves x 16 rows; key tiles
// of 32; online softmax; WMMA for QK^T and PV. K tile async, double-buffered.
__global__ __launch_bounds__(256) void attn_k(const u16* __restrict__ q,
                                              const u16* __restrict__ k,
                                              const u16* __restrict__ v,
                                              u16* __restrict__ outp) {
  __shared__ __align__(16) u16 sK[2][32 * 64];     // [key][dim]
  __shared__ __align__(16) u16 sVT[2][64 * 32];    // [dim][key]
  __shared__ __align__(16) u16 sP[8][16 * 32];     // per-wave P scratch
  const int bh = blockIdx.y;
  const int wave = threadIdx.x >> 5, lane = threadIdx.x & 31;
  const int hf = lane >> 4, ln = lane & 15;
  const int qr0 = blockIdx.x * 128 + wave * 16;
  const u16* qb = q + (size_t)bh * kSeq * kHD;
  const u16* kb = k + (size_t)bh * kSeq * kHD;
  const u16* vb = v + (size_t)bh * kSeq * kHD;

  Frag qa[2];
  {
    const u16* p = qb + (size_t)(qr0 + ln) * kHD;
    qa[0].u[0] = *(const uint4*)(p + 8 * hf);
    qa[0].u[1] = *(const uint4*)(p + 16 + 8 * hf);
    qa[1].u[0] = *(const uint4*)(p + 32 + 8 * hf);
    qa[1].u[1] = *(const uint4*)(p + 48 + 8 * hf);
  }
  const v8f zz = {};
  v8f o[4];
#pragma unroll
  for (int t4 = 0; t4 < 4; t4++) o[t4] = zz;
  float mr[8], lr[8];
#pragma unroll
  for (int r = 0; r < 8; r++) { mr[r] = -1e30f; lr[r] = 0.f; }

  auto stageKV = [&](int kt, int buf) {
    const int key = threadIdx.x >> 3, d8 = (threadIdx.x & 7) * 8;
    async_ld_b128(&sK[buf][key * 64 + d8], &kb[(size_t)(kt * 32 + key) * kHD + d8]);
    uint4 vv = *(const uint4*)&vb[(size_t)(kt * 32 + key) * kHD + d8];
    const u16* vs = (const u16*)&vv;
#pragma unroll
    for (int i = 0; i < 8; i++) sVT[buf][(d8 + i) * 32 + key] = vs[i];
  };

  const int nkt = (blockIdx.x + 1) * 4;   // causal: keys up to end of block
  stageKV(0, 0);
  for (int kt = 0; kt < nkt; ++kt) {
    const int cur = kt & 1;
    if (kt + 1 < nkt) { stageKV(kt + 1, cur ^ 1); wait_async<1>(); }
    else wait_async<0>();
    __syncthreads();

    if (kt * 32 <= qr0 + 15) {   // wave not fully masked
      v8f sc[2];
#pragma unroll
      for (int g = 0; g < 2; ++g) {
        Frag b0, b1;
        const u16* kp = &sK[cur][(g * 16 + ln) * 64];
        b0.u[0] = *(const uint4*)(kp + 8 * hf);
        b0.u[1] = *(const uint4*)(kp + 8 * hf + 16);
        b1.u[0] = *(const uint4*)(kp + 32 + 8 * hf);
        b1.u[1] = *(const uint4*)(kp + 32 + 8 * hf + 16);
        v8f z = zz;
        z = wmma_bf16(qa[0].bf, b0.bf, z);
        z = wmma_bf16(qa[1].bf, b1.bf, z);
        sc[g] = z;
      }
      float pw[2][8], alpha[8];
#pragma unroll
      for (int r = 0; r < 8; r++) {
        const int qi = qr0 + r + 8 * hf;
        const int k0i = kt * 32 + ln;
        float v0 = (k0i <= qi) ? sc[0][r] * 0.125f : -1e30f;
        float v1 = (k0i + 16 <= qi) ? sc[1][r] * 0.125f : -1e30f;
        float t = fmaxf(v0, v1);
#pragma unroll
        for (int msk = 8; msk >= 1; msk >>= 1) t = fmaxf(t, __shfl_xor(t, msk));
        const float mn = fmaxf(mr[r], t);
        const float a = __expf(mr[r] - mn);
        const float p0 = __expf(v0 - mn), p1 = __expf(v1 - mn);
        float sm = p0 + p1;
#pragma unroll
        for (int msk = 8; msk >= 1; msk >>= 1) sm += __shfl_xor(sm, msk);
        lr[r] = lr[r] * a + sm;
        mr[r] = mn; alpha[r] = a;
        pw[0][r] = p0; pw[1][r] = p1;
      }
#pragma unroll
      for (int g = 0; g < 2; g++)
#pragma unroll
        for (int r = 0; r < 8; r++)
          sP[wave][(r + 8 * hf) * 32 + g * 16 + ln] = f2bf(pw[g][r]);
#pragma unroll
      for (int t4 = 0; t4 < 4; t4++)
#pragma unroll
        for (int r = 0; r < 8; r++) o[t4][r] *= alpha[r];
      Frag pa;
      {
        const u16* pp = &sP[wave][ln * 32 + 8 * hf];
        pa.u[0] = *(const uint4*)pp;
        pa.u[1] = *(const uint4*)(pp + 16);
      }
#pragma unroll
      for (int t4 = 0; t4 < 4; t4++) {
        Frag vf;
        const u16* vp = &sVT[cur][(t4 * 16 + ln) * 32 + 8 * hf];
        vf.u[0] = *(const uint4*)vp;
        vf.u[1] = *(const uint4*)(vp + 16);
        o[t4] = wmma_bf16(pa.bf, vf.bf, o[t4]);
      }
    }
    __syncthreads();
  }

  const int b = bh / kNH, hh = bh % kNH;
#pragma unroll
  for (int r = 0; r < 8; r++) {
    const float rl = fast_rcp(lr[r]);
#pragma unroll
    for (int t4 = 0; t4 < 4; t4++) {
      const int qi = qr0 + r + 8 * hf;
      const int col = hh * kHD + t4 * 16 + ln;
      outp[(size_t)(b * kSeq + qi) * kH + col] = f2bf(o[t4][r] * rl);
    }
  }
}

// ---------------------------------------------------------------- d_out init:
// main region = attention residual (x2); aux slot zeroed for atomics.
__global__ __launch_bounds__(256) void init_out_k(const float* __restrict__ x2,
                                                  float* __restrict__ outp) {
  const size_t i = (size_t)blockIdx.x * 256 + threadIdx.x;
  outp[i] = x2[i];
  if (i == 0) outp[(size_t)kNT * kH] = 0.f;
}

// ---------------------------------------------------------------- Gating:
// logits = xn2 @ gate_w, softmax, top-2 normalized weights, z-loss aux.
__global__ __launch_bounds__(256) void gate_k(const u16* __restrict__ xn,
                                              const float* __restrict__ gw,
                                              float* __restrict__ comb,
                                              float* __restrict__ aux) {
  const int tok = blockIdx.x;
  const int wave = threadIdx.x >> 5, lane = threadIdx.x & 31;   // wave = expert
  float s = 0.f;
  for (int i = lane; i < kH; i += 32)
    s += bf2f(xn[(size_t)tok * kH + i]) * gw[(size_t)i * kE + wave];
#pragma unroll
  for (int m = 16; m >= 1; m >>= 1) s += __shfl_xor(s, m);
  __shared__ float lg[kE];
  if (lane == 0) lg[wave] = s;
  __syncthreads();
  if (threadIdx.x == 0) {
    float mx = lg[0];
    for (int e = 1; e < kE; e++) mx = fmaxf(mx, lg[e]);
    float p[kE]; float se = 0.f;
    for (int e = 0; e < kE; e++) { p[e] = __expf(lg[e] - mx); se += p[e]; }
    int i0 = 0;
    for (int e = 1; e < kE; e++) if (p[e] > p[i0]) i0 = e;
    int i1 = (i0 == 0) ? 1 : 0;
    for (int e = 0; e < kE; e++) if (e != i0 && p[e] > p[i1]) i1 = e;
    const float rw = fast_rcp(p[i0] + p[i1]);
    for (int e = 0; e < kE; e++) comb[(size_t)tok * kE + e] = 0.f;
    comb[(size_t)tok * kE + i0] = p[i0] * rw;
    comb[(size_t)tok * kE + i1] = p[i1] * rw;
    const float lse = mx + __logf(se);
    atomicAdd(aux, 0.001f * lse * lse * (1.f / kNT));
  }
}

extern "C" void kernel_launch(void* const* d_in, const int* in_sizes, int n_in,
                              void* d_out, int out_size, void* d_ws, size_t ws_size,
                              hipStream_t stream) {
  (void)in_sizes; (void)n_in; (void)out_size; (void)ws_size;
  const float* hidden = (const float*)d_in[0];
  // d_in[1] = attention_mask (causal; computed analytically)
  const float* wq     = (const float*)d_in[2];
  const float* wk     = (const float*)d_in[3];
  const float* wv     = (const float*)d_in[4];
  const float* wo     = (const float*)d_in[5];
  const float* ln1    = (const float*)d_in[6];
  const float* ln2    = (const float*)d_in[7];
  const float* gate_w = (const float*)d_in[8];
  const float* w_gate = (const float*)d_in[9];
  const float* w_up   = (const float*)d_in[10];
  const float* w_down = (const float*)d_in[11];
  float* outp = (float*)d_out;

  // -------- workspace carve-up
  char* p = (char*)d_ws;
  auto alloc = [&](size_t bytes) { char* r = p; p += (bytes + 255) & ~(size_t)255; return r; };
  const size_t actB = (size_t)kNT * kH * 2;          // bf16 [4096,768]
  u16*   xn1   = (u16*)alloc(actB);
  u16*   qlin  = (u16*)alloc(actB);
  u16*   klin  = (u16*)alloc(actB);
  u16*   vlin  = (u16*)alloc(actB);
  u16*   qatt  = (u16*)alloc(actB);
  u16*   katt  = (u16*)alloc(actB);
  u16*   vatt  = (u16*)alloc(actB);
  u16*   attnb = (u16*)alloc(actB);
  float* x2    = (float*)alloc((size_t)kNT * kH * 4);
  u16*   xn2   = (u16*)alloc(actB);
  float* comb  = (float*)alloc((size_t)kNT * kE * 4);
  u16*   hbuf  = (u16*)alloc((size_t)kNT * kI * 2);
  u16*   wqT   = (u16*)alloc((size_t)kH * kH * 2);   // bf16 [N=768][K=768]
  u16*   wkT   = (u16*)alloc((size_t)kH * kH * 2);
  u16*   wvT   = (u16*)alloc((size_t)kH * kH * 2);
  u16*   woT   = (u16*)alloc((size_t)kH * kH * 2);
  u16*   wgT   = (u16*)alloc((size_t)kE * kI * kH * 2);  // per-e [N=2048][K=768]
  u16*   wuT   = (u16*)alloc((size_t)kE * kI * kH * 2);
  u16*   wdT   = (u16*)alloc((size_t)kE * kH * kI * 2);  // per-e [N=768][K=2048]

  // -------- one-time weight convert+transpose (fp32 -> bf16, N-major)
  const dim3 gHH(kH / 64, kH / 64);       // (12,12)
  transpose_bf16_k<<<gHH, 256, 0, stream>>>(wq, wqT, kH, kH);
  transpose_bf16_k<<<gHH, 256, 0, stream>>>(wk, wkT, kH, kH);
  transpose_bf16_k<<<gHH, 256, 0, stream>>>(wv, wvT, kH, kH);
  transpose_bf16_k<<<gHH, 256, 0, stream>>>(wo, woT, kH, kH);
  for (int e = 0; e < kE; ++e) {
    transpose_bf16_k<<<dim3(kH / 64, kI / 64), 256, 0, stream>>>(
        w_gate + (size_t)e * kH * kI, wgT + (size_t)e * kI * kH, kH, kI);
    transpose_bf16_k<<<dim3(kH / 64, kI / 64), 256, 0, stream>>>(
        w_up + (size_t)e * kH * kI, wuT + (size_t)e * kI * kH, kH, kI);
    transpose_bf16_k<<<dim3(kI / 64, kH / 64), 256, 0, stream>>>(
        w_down + (size_t)e * kI * kH, wdT + (size_t)e * kH * kI, kI, kH);
  }

  const dim3 gP(kNT / 128, kH / 64);   // (32,12) for H-wide GEMMs

  rmsnorm_k<<<kNT, 256, 0, stream>>>(hidden, ln1, xn1);
  gemm_k<0><<<gP, 256, 0, stream>>>(xn1, wqT, qlin, nullptr, nullptr, kNT, kH, kH);
  gemm_k<0><<<gP, 256, 0, stream>>>(xn1, wkT, klin, nullptr, nullptr, kNT, kH, kH);
  gemm_k<0><<<gP, 256, 0, stream>>>(xn1, wvT, vlin, nullptr, nullptr, kNT, kH, kH);
  rope_k<<<(kNT * kH) / 256, 256, 0, stream>>>(qlin, klin, vlin, qatt, katt, vatt);
  attn_k<<<dim3(kSeq / 128, kB * kNH), 256, 0, stream>>>(qatt, katt, vatt, attnb);
  gemm_k<1><<<gP, 256, 0, stream>>>(attnb, woT, nullptr, x2, hidden, kNT, kH, kH);
  rmsnorm_k<<<kNT, 256, 0, stream>>>(x2, ln2, xn2);
  init_out_k<<<(kNT * kH) / 256, 256, 0, stream>>>(x2, outp);
  gate_k<<<kNT, 256, 0, stream>>>(xn2, gate_w, comb, outp + (size_t)kNT * kH);
  for (int e = 0; e < kE; ++e) {
    gemm_gated_k<<<dim3(kNT / 128, kI / 64), 256, 0, stream>>>(
        xn2, wgT + (size_t)e * kI * kH, wuT + (size_t)e * kI * kH, hbuf, kNT, kI, kH);
    gemm_k<3><<<gP, 256, 0, stream>>>(
        hbuf, wdT + (size_t)e * kH * kI, nullptr, outp, comb + e, kNT, kH, kI);
  }
}
